// motion_ar_17463337026116
// MI455X (gfx1250) — matrix operand order repaired and checked by reading, not compiled, not run on hardware
//
#include <hip/hip_runtime.h>
#include <math.h>

// ---- problem constants (from reference) ----
#define Bz 256
#define Tz 64
#define Hz 512
#define Kz 1024
#define Sz 256
#define NSTEP 63                       // steps 0..62
#define HEADSTRIDE (256ull*63ull*1024ull)
#define OUT_FX 65536ull                // sampled_delta is [B,64,4] = 65536 floats at offset 0

typedef __bf16 bf16;
typedef __attribute__((ext_vector_type(16))) __bf16 v16bf;
typedef __attribute__((ext_vector_type(8)))  __bf16 v8bf;
typedef __attribute__((ext_vector_type(8)))  float  v8f;

// A fragment (16x32 bf16): lane holds 16 contiguous K = k0+half*16 .. +15.
__device__ __forceinline__ v16bf load_afrag(const bf16* __restrict__ A, int lda,
                                            int row, int k0, int half) {
    const bf16* ap = A + (size_t)row * lda + k0 + (half << 4);
    v8bf lo = *(const v8bf*)(ap);        // K = k0+half*16 .. +7
    v8bf hi = *(const v8bf*)(ap + 8);    // K = k0+half*16+8 .. +15
    v16bf a;
#pragma unroll
    for (int e = 0; e < 8; ++e) { a[e] = lo[e]; a[e + 8] = hi[e]; }
    return a;
}

#define WMMA_BF16(A_, B_, C_) \
    __builtin_amdgcn_wmma_f32_16x16x32_bf16(false, (A_), false, (B_), (short)0, (C_), false, false)

// =====================================================================
// One-wave-per-32x64-tile bf16 WMMA GEMM, software-pipelined K loop:
//   out[M,N] = act( A1[M,K1] @ W[:, 0:K1]ᵀ + A2[M,K2] @ W[:, K1:K1+K2]ᵀ
//                   + bias + addend )
// W stored row-major [N, K1+K2] bf16. headMode=1 scatters N=4096 into
// 4 head slices of d_out at `step`.
// =====================================================================
__global__ __launch_bounds__(32)
void gemm_wmma_bf16(const bf16* __restrict__ A1, int lda1, int K1,
                    const bf16* __restrict__ A2, int lda2, int K2,
                    const bf16* __restrict__ W,  int ldw,
                    const float* __restrict__ bias,
                    const float* __restrict__ addend, int ldadd,
                    float* __restrict__ Cf, int ldc,
                    bf16* __restrict__ Cb, int ldcb,
                    int relu, int headMode, int step)
{
    const int lane = threadIdx.x;
    const int half = lane >> 4;      // 0/1
    const int lr   = lane & 15;      // row (A) or col (B/C) within tile
    const int m0   = blockIdx.x * 32;
    const int n0   = blockIdx.y * 64;
    const int rA0  = m0 + lr;        // A rows for M-tile 0
    const int rA1  = m0 + 16 + lr;   // A rows for M-tile 1

    v8f acc[8];
#pragma unroll
    for (int i = 0; i < 8; ++i) acc[i] = (v8f){};

    for (int seg = 0; seg < 2; ++seg) {
        const bf16* A = seg ? A2 : A1;
        const int lda = seg ? lda2 : lda1;
        const int K   = seg ? K2  : K1;
        const int wof = seg ? K1  : 0;
        if (K == 0 || A == nullptr) continue;

        const bf16* wrow = W + (size_t)(n0 + lr) * ldw + wof + (half << 4);

        // ---- pipeline prologue: fragments for k0 = 0 ----
        v16bf a0 = load_afrag(A, lda, rA0, 0, half);
        v16bf a1 = load_afrag(A, lda, rA1, 0, half);
        v16bf b0 = *(const v16bf*)(wrow);
        v16bf b1 = *(const v16bf*)(wrow + (size_t)16 * ldw);
        v16bf b2 = *(const v16bf*)(wrow + (size_t)32 * ldw);
        v16bf b3 = *(const v16bf*)(wrow + (size_t)48 * ldw);

        for (int k0 = 0; k0 + 32 < K; k0 += 32) {
            const int kn = k0 + 32;
            // issue next K-step's loads before current WMMAs (overlap)
            v16bf na0 = load_afrag(A, lda, rA0, kn, half);
            v16bf na1 = load_afrag(A, lda, rA1, kn, half);
            v16bf nb0 = *(const v16bf*)(wrow + kn);
            v16bf nb1 = *(const v16bf*)(wrow + kn + (size_t)16 * ldw);
            v16bf nb2 = *(const v16bf*)(wrow + kn + (size_t)32 * ldw);
            v16bf nb3 = *(const v16bf*)(wrow + kn + (size_t)48 * ldw);

            acc[0] = WMMA_BF16(a0, b0, acc[0]);
            acc[1] = WMMA_BF16(a0, b1, acc[1]);
            acc[2] = WMMA_BF16(a0, b2, acc[2]);
            acc[3] = WMMA_BF16(a0, b3, acc[3]);
            acc[4] = WMMA_BF16(a1, b0, acc[4]);
            acc[5] = WMMA_BF16(a1, b1, acc[5]);
            acc[6] = WMMA_BF16(a1, b2, acc[6]);
            acc[7] = WMMA_BF16(a1, b3, acc[7]);

            a0 = na0; a1 = na1; b0 = nb0; b1 = nb1; b2 = nb2; b3 = nb3;
        }
        // ---- pipeline epilogue: last K-step ----
        acc[0] = WMMA_BF16(a0, b0, acc[0]);
        acc[1] = WMMA_BF16(a0, b1, acc[1]);
        acc[2] = WMMA_BF16(a0, b2, acc[2]);
        acc[3] = WMMA_BF16(a0, b3, acc[3]);
        acc[4] = WMMA_BF16(a1, b0, acc[4]);
        acc[5] = WMMA_BF16(a1, b1, acc[5]);
        acc[6] = WMMA_BF16(a1, b2, acc[6]);
        acc[7] = WMMA_BF16(a1, b3, acc[7]);
    }

    // ---- epilogue: bias / addend / relu / store ----
#pragma unroll
    for (int mt = 0; mt < 2; ++mt) {
#pragma unroll
        for (int nt = 0; nt < 4; ++nt) {
            const v8f a = acc[mt * 4 + nt];
            const int n = n0 + nt * 16 + lr;
            const float bn = bias ? bias[n] : 0.0f;
#pragma unroll
            for (int r = 0; r < 8; ++r) {
                const int m = m0 + mt * 16 + r + 8 * half;  // C layout: lanes 16-31 -> M+8
                float v = a[r] + bn;
                if (addend) v += addend[(size_t)m * ldadd + n];
                if (relu)   v = v > 0.0f ? v : 0.0f;
                if (headMode) {
                    const int head = n >> 10, col = n & 1023;
                    Cf[(size_t)head * HEADSTRIDE + ((size_t)m * NSTEP + step) * Kz + col] = v;
                } else if (Cf) {
                    Cf[(size_t)m * ldc + n] = v;
                }
                if (Cb) Cb[(size_t)m * ldcb + n] = (bf16)v;
            }
        }
    }
}

// ---- f32 -> bf16 strided conversion (also used for concat packing) ----
__global__ void k_cvt_bf16(const float* __restrict__ src, bf16* __restrict__ dst,
                           long rows, long cols, long dstStride, long dstOff) {
    long i = (long)blockIdx.x * blockDim.x + threadIdx.x;
    if (i >= rows * cols) return;
    long r = i / cols, c = i - r * cols;
    dst[r * dstStride + dstOff + c] = (bf16)src[i];
}

__global__ void k_copy_f32(const float* __restrict__ src, float* __restrict__ dst, long n) {
    long i = (long)blockIdx.x * blockDim.x + threadIdx.x;
    if (i < n) dst[i] = src[i];
}

__global__ void k_add_f32(const float* __restrict__ a, const float* __restrict__ b,
                          float* __restrict__ dst, long n) {
    long i = (long)blockIdx.x * blockDim.x + threadIdx.x;
    if (i < n) dst[i] = a[i] + b[i];
}

// ---- fc1 + shortcut of the residual block (K=4, scalar f32) ----
__global__ void k_fc1(const float* __restrict__ x, const float* __restrict__ W1,
                      const float* __restrict__ b1, const float* __restrict__ Ws,
                      const float* __restrict__ bs,
                      bf16* __restrict__ h1, float* __restrict__ sc) {
    int i = blockIdx.x * blockDim.x + threadIdx.x;   // B*H
    int b = i >> 9, j = i & 511;
    const float* xb = x + b * 4;
    float a0 = W1[j*4+0]*xb[0] + W1[j*4+1]*xb[1] + W1[j*4+2]*xb[2] + W1[j*4+3]*xb[3] + b1[j];
    h1[i] = (bf16)(a0 > 0.0f ? a0 : 0.0f);
    sc[i] = Ws[j*4+0]*xb[0] + Ws[j*4+1]*xb[1] + Ws[j*4+2]*xb[2] + Ws[j*4+3]*xb[3] + bs[j];
}

// ---- LSTM pointwise + residual-around-LSTM ----
__device__ __forceinline__ float sigf(float x) { return 1.0f / (1.0f + expf(-x)); }

__global__ void k_lstm(const float* __restrict__ gates, float* __restrict__ c,
                       float* __restrict__ h, bf16* __restrict__ hb,
                       const float* __restrict__ xe, bf16* __restrict__ xob) {
    int i = blockIdx.x * blockDim.x + threadIdx.x;   // B*H
    int b = i >> 9, j = i & 511;
    const float* g = gates + (size_t)b * (4 * Hz);
    float ig = g[j], fg = g[Hz + j], gg = g[2 * Hz + j], og = g[3 * Hz + j];
    float cn = sigf(fg) * c[i] + sigf(ig) * tanhf(gg);
    float hn = sigf(og) * tanhf(cn);
    c[i] = cn;
    h[i] = hn;
    hb[i] = (bf16)hn;
    xob[i] = (bf16)(hn + xe[i]);   // xo = h2 + xe
}

// ---- step-0 input: x = obs[:,0,:] ----
__global__ void k_x0(const float* __restrict__ obs, float* __restrict__ x) {
    int i = blockIdx.x * blockDim.x + threadIdx.x;
    if (i >= Bz * 4) return;
    int b = i >> 2, j = i & 3;
    x[i] = obs[((size_t)b * Tz) * 4 + j];
}

// ---- sampled_delta edges: [:,0] and [:,T-1] = observation ----
__global__ void k_edge(const float* __restrict__ obs, float* __restrict__ out_sd) {
    int i = blockIdx.x * blockDim.x + threadIdx.x;
    if (i >= Bz * 4) return;
    int b = i >> 2, j = i & 3;
    out_sd[((size_t)b * Tz + 0)      * 4 + j] = obs[((size_t)b * Tz + 0)      * 4 + j];
    out_sd[((size_t)b * Tz + Tz - 1) * 4 + j] = obs[((size_t)b * Tz + Tz - 1) * 4 + j];
}

// ---- per-step input selection: argmax over K per head, gather centroids ----
__global__ __launch_bounds__(128)
void k_input(const float* __restrict__ out_logits,   // d_out + OUT_FX
             const float* __restrict__ cx, const float* __restrict__ cy,
             const float* __restrict__ cw, const float* __restrict__ ch,
             const float* __restrict__ obs, const int* __restrict__ maskp,
             float* __restrict__ x, float* __restrict__ out_sd, int s) {
    const int b = blockIdx.x, tid = threadIdx.x;
    __shared__ float sval[128];
    __shared__ int   sidx[128];
    __shared__ float samp[4];
    const float* cents[4] = {cx, cy, cw, ch};
    for (int q = 0; q < 4; ++q) {
        const float* L = out_logits + (size_t)q * HEADSTRIDE + ((size_t)b * NSTEP + (s - 1)) * Kz;
        float best = -3.4e38f; int bi = 0;
        for (int k = tid; k < Kz; k += 128) {
            float v = L[k];
            if (v > best) { best = v; bi = k; }
        }
        sval[tid] = best; sidx[tid] = bi; __syncthreads();
        for (int off = 64; off > 0; off >>= 1) {
            if (tid < off) {
                if (sval[tid+off] > sval[tid] ||
                    (sval[tid+off] == sval[tid] && sidx[tid+off] < sidx[tid])) {
                    sval[tid] = sval[tid+off]; sidx[tid] = sidx[tid+off];
                }
            }
            __syncthreads();
        }
        if (tid == 0) samp[q] = cents[q][sidx[0]];
        __syncthreads();
    }
    if (tid < 4) {
        int j = tid;
        float v = (s < *maskp) ? obs[((size_t)b * Tz + s) * 4 + j] : samp[j];
        x[b * 4 + j] = v;
        out_sd[((size_t)b * Tz + s) * 4 + j] = v;
    }
}

#define CDIV(a, b) (((a) + (b) - 1) / (b))

extern "C" void kernel_launch(void* const* d_in, const int* in_sizes, int n_in,
                              void* d_out, int out_size, void* d_ws, size_t ws_size,
                              hipStream_t stream) {
    (void)in_sizes; (void)n_in; (void)out_size; (void)ws_size;
    const float* obs   = (const float*)d_in[0];
    const float* soc   = (const float*)d_in[1];
    const int*   maskp = (const int*)  d_in[2];
    const float* cx    = (const float*)d_in[3];
    const float* cy    = (const float*)d_in[4];
    const float* cw    = (const float*)d_in[5];
    const float* ch    = (const float*)d_in[6];
    const float* Wrb1  = (const float*)d_in[7];
    const float* brb1  = (const float*)d_in[8];
    const float* Wrb2  = (const float*)d_in[9];
    const float* brb2  = (const float*)d_in[10];
    const float* Wrbs  = (const float*)d_in[11];
    const float* brbs  = (const float*)d_in[12];
    const float* Wemb  = (const float*)d_in[13];
    const float* bemb  = (const float*)d_in[14];
    const float* Wih   = (const float*)d_in[15];
    const float* bih   = (const float*)d_in[16];
    const float* Whh   = (const float*)d_in[17];
    const float* bhh   = (const float*)d_in[18];
    const float* Wfx   = (const float*)d_in[19];
    const float* bfx   = (const float*)d_in[20];
    const float* Wfy   = (const float*)d_in[21];
    const float* bfy   = (const float*)d_in[22];
    const float* Wfw   = (const float*)d_in[23];
    const float* bfw   = (const float*)d_in[24];
    const float* Wfh   = (const float*)d_in[25];
    const float* bfh   = (const float*)d_in[26];
    const float* h0    = (const float*)d_in[27];
    const float* c0    = (const float*)d_in[28];
    float* out = (float*)d_out;

    // ---- workspace layout (bytes, 256-aligned) ----
    char* ws = (char*)d_ws;
    size_t o = 0;
    auto take = [&](size_t bytes) { size_t r = o; o = (o + bytes + 255) & ~(size_t)255; return r; };
    bf16*  WRB2p  = (bf16*)(ws + take((size_t)Hz * Hz * 2));
    bf16*  WEMBp  = (bf16*)(ws + take((size_t)Hz * (Hz + Sz) * 2));
    bf16*  WIHHp  = (bf16*)(ws + take((size_t)4 * Hz * (2 * Hz) * 2));
    bf16*  WHEADp = (bf16*)(ws + take((size_t)4 * Kz * Hz * 2));
    bf16*  SOCp   = (bf16*)(ws + take((size_t)Bz * Tz * Sz * 2));
    float* BSUMp  = (float*)(ws + take((size_t)4 * Hz * 4));
    float* BHEADp = (float*)(ws + take((size_t)4 * Kz * 4));
    float* Xp     = (float*)(ws + take((size_t)Bz * 4 * 4));
    bf16*  H1p    = (bf16*)(ws + take((size_t)Bz * Hz * 2));
    float* SCp    = (float*)(ws + take((size_t)Bz * Hz * 4));
    float* XEFp   = (float*)(ws + take((size_t)Bz * Hz * 4));
    bf16*  XEBp   = (bf16*)(ws + take((size_t)Bz * Hz * 2));
    bf16*  XSBp   = (bf16*)(ws + take((size_t)Bz * Hz * 2));
    float* GATESp = (float*)(ws + take((size_t)Bz * 4 * Hz * 4));
    float* HFp    = (float*)(ws + take((size_t)Bz * Hz * 4));
    bf16*  HBp    = (bf16*)(ws + take((size_t)Bz * Hz * 2));
    float* CFp    = (float*)(ws + take((size_t)Bz * Hz * 4));
    bf16*  XOBp   = (bf16*)(ws + take((size_t)Bz * Hz * 2));

    // ---- one-time weight / state prep ----
    k_cvt_bf16<<<CDIV(Hz*Hz,256),256,0,stream>>>(Wrb2, WRB2p, Hz, Hz, Hz, 0);
    k_cvt_bf16<<<CDIV(Hz*(Hz+Sz),256),256,0,stream>>>(Wemb, WEMBp, Hz, Hz+Sz, Hz+Sz, 0);
    // gates weight concat along K: [W_ih | W_hh] -> [4H, 2H]
    k_cvt_bf16<<<CDIV(4*Hz*Hz,256),256,0,stream>>>(Wih, WIHHp, 4*Hz, Hz, 2*Hz, 0);
    k_cvt_bf16<<<CDIV(4*Hz*Hz,256),256,0,stream>>>(Whh, WIHHp, 4*Hz, Hz, 2*Hz, Hz);
    // heads weight concat along N: [4K, H]
    k_cvt_bf16<<<CDIV(Kz*Hz,256),256,0,stream>>>(Wfx, WHEADp + 0ull*Kz*Hz, Kz, Hz, Hz, 0);
    k_cvt_bf16<<<CDIV(Kz*Hz,256),256,0,stream>>>(Wfy, WHEADp + 1ull*Kz*Hz, Kz, Hz, Hz, 0);
    k_cvt_bf16<<<CDIV(Kz*Hz,256),256,0,stream>>>(Wfw, WHEADp + 2ull*Kz*Hz, Kz, Hz, Hz, 0);
    k_cvt_bf16<<<CDIV(Kz*Hz,256),256,0,stream>>>(Wfh, WHEADp + 3ull*Kz*Hz, Kz, Hz, Hz, 0);
    // social -> bf16 (whole [B,T,S] tensor)
    k_cvt_bf16<<<CDIV((long)Bz*Tz*Sz,256),256,0,stream>>>(soc, SOCp, (long)Bz*Tz, Sz, Sz, 0);
    // combined biases
    k_add_f32<<<CDIV(4*Hz,256),256,0,stream>>>(bih, bhh, BSUMp, 4*Hz);
    k_copy_f32<<<CDIV(Kz,256),256,0,stream>>>(bfx, BHEADp + 0*Kz, Kz);
    k_copy_f32<<<CDIV(Kz,256),256,0,stream>>>(bfy, BHEADp + 1*Kz, Kz);
    k_copy_f32<<<CDIV(Kz,256),256,0,stream>>>(bfw, BHEADp + 2*Kz, Kz);
    k_copy_f32<<<CDIV(Kz,256),256,0,stream>>>(bfh, BHEADp + 3*Kz, Kz);
    // initial state
    k_copy_f32<<<CDIV(Bz*Hz,256),256,0,stream>>>(h0, HFp, Bz*Hz);
    k_copy_f32<<<CDIV(Bz*Hz,256),256,0,stream>>>(c0, CFp, Bz*Hz);
    k_cvt_bf16<<<CDIV(Bz*Hz,256),256,0,stream>>>(h0, HBp, Bz, Hz, Hz, 0);
    // sampled_delta edges
    k_edge<<<CDIV(Bz*4,256),256,0,stream>>>(obs, out);

    // ---- 63 sequential timesteps ----
    for (int s = 0; s < NSTEP; ++s) {
        if (s == 0)
            k_x0<<<CDIV(Bz*4,256),256,0,stream>>>(obs, Xp);
        else
            k_input<<<Bz,128,0,stream>>>(out + OUT_FX, cx, cy, cw, ch, obs, maskp, Xp, out, s);

        // residual block fc1 + shortcut
        k_fc1<<<CDIV(Bz*Hz,256),256,0,stream>>>(Xp, Wrb1, brb1, Wrbs, brbs, H1p, SCp);
        // xe = relu(h1 @ W_rb2ᵀ + b_rb2 + sc)  -> f32 + bf16
        gemm_wmma_bf16<<<dim3(Bz/32, Hz/64),32,0,stream>>>(
            H1p, Hz, Hz, nullptr, 0, 0, WRB2p, Hz, brb2,
            SCp, Hz, XEFp, Hz, XEBp, Hz, /*relu*/1, 0, 0);
        // xs = relu([xe, social_t] @ W_embᵀ + b_emb) -> bf16
        gemm_wmma_bf16<<<dim3(Bz/32, Hz/64),32,0,stream>>>(
            XEBp, Hz, Hz, SOCp + (size_t)s * Sz, Tz * Sz, Sz, WEMBp, Hz + Sz, bemb,
            nullptr, 0, nullptr, 0, XSBp, Hz, /*relu*/1, 0, 0);
        // gates = [xs, h] @ [W_ih|W_hh]ᵀ + (b_ih + b_hh) -> f32
        gemm_wmma_bf16<<<dim3(Bz/32, (4*Hz)/64),32,0,stream>>>(
            XSBp, Hz, Hz, HBp, Hz, Hz, WIHHp, 2*Hz, BSUMp,
            nullptr, 0, GATESp, 4*Hz, nullptr, 0, /*relu*/0, 0, 0);
        // LSTM pointwise; xo = h2 + xe
        k_lstm<<<CDIV(Bz*Hz,256),256,0,stream>>>(GATESp, CFp, HFp, HBp, XEFp, XOBp);
        // head logits, scattered into d_out per-head slices at step s
        gemm_wmma_bf16<<<dim3(Bz/32, (4*Kz)/64),32,0,stream>>>(
            XOBp, Hz, Hz, nullptr, 0, 0, WHEADp, Hz, BHEADp,
            nullptr, 0, out + OUT_FX, 0, nullptr, 0, /*relu*/0, /*headMode*/1, s);
    }
}
